// PeepholeCell_46677704573636
// MI455X (gfx1250) — matrix operand order repaired and checked by reading, not compiled, not run on hardware
//
#include <hip/hip_runtime.h>

typedef __attribute__((ext_vector_type(16))) _Float16 v16h;
typedef __attribute__((ext_vector_type(8)))  _Float16 v8h;
typedef __attribute__((ext_vector_type(2)))  _Float16 h2;
typedef __attribute__((ext_vector_type(8)))  float    v8f;

#define BROWS 16384
#define DIN   512
#define DH    512
#define MT    128   // block tile M (batch rows)
#define NT    64    // block tile N (hidden cols)
#define KT    32    // k-step per WMMA pass
#define LDA   40    // padded halves per A row (bank-spread)
#define LDB   40    // padded halves per B n-row
#define NSTEP 32    // 2 phases * (512/32)

__device__ __forceinline__ h2 pkh(float a, float b) {
    auto t = __builtin_amdgcn_cvt_pkrtz(a, b);   // v_cvt_pk_f16_f32 (RTZ)
    h2 r; r.x = (_Float16)t.x; r.y = (_Float16)t.y;
    return r;
}

__device__ __forceinline__ float sigmoid_fast(float x) {
    return __builtin_amdgcn_rcpf(1.0f + __expf(-x));
}

__device__ __forceinline__ float tanh_fast(float x) {
    float e = __expf(fminf(-2.0f * x, 80.0f));
    return (1.0f - e) * __builtin_amdgcn_rcpf(1.0f + e);
}

__global__ __launch_bounds__(256)
void lstm_fused_wmma(const float* __restrict__ X,
                     const float* __restrict__ Cprev,
                     const float* __restrict__ Wf, const float* __restrict__ Wi,
                     const float* __restrict__ Wo, const float* __restrict__ Wc,
                     const float* __restrict__ Uf, const float* __restrict__ Ui,
                     const float* __restrict__ Uo, const float* __restrict__ Uc,
                     const float* __restrict__ bf, const float* __restrict__ bi,
                     const float* __restrict__ bo, const float* __restrict__ bc,
                     float* __restrict__ H)
{
    __shared__ _Float16 lA[MT][LDA];        // X / c_prev tile, row-major [m][k]
    __shared__ _Float16 lB[4][NT][LDB];     // weight tiles, transposed [gate][n][k]

    const int tid  = threadIdx.x;
    const int lane = tid & 31;
    const int wave = tid >> 5;              // 0..7 -> 16-row strip of the 128-row tile
    const int m0   = blockIdx.y * MT;
    const int n0   = blockIdx.x * NT;

    const int nl    = lane & 15;            // fragment row/col index
    const int khalf = lane >> 4;            // lane half selects K sub-range

    // flattened accumulators: t = gate*4 + n_subtile
    v8f acc[16];
    #pragma unroll
    for (int t = 0; t < 16; ++t)
        #pragma unroll
        for (int e = 0; e < 8; ++e)
            acc[t][e] = 0.0f;

    const float* Bp[8] = { Wf, Wi, Wo, Wc, Uf, Ui, Uo, Uc };

    // per-thread staging coordinates (constant across steps)
    const int arow = tid >> 3;              // A: rows arow, arow+32, +64, +96
    const int akq  = tid & 7;               // A: float4 slot along k
    const int kp   = tid >> 4;              // B: k-pair index 0..15 (k = 2*kp)
    const int nq   = tid & 15;              // B: float4 slot along n

    float4 rA[4], rB0[4], rB1[4];           // global prefetch registers

#define ISSUE(s)                                                                   \
    do {                                                                           \
        const float*        As_ = ((s) < 16) ? X : Cprev;                          \
        const float* const* Ws_ = Bp + (((s) < 16) ? 0 : 4);                       \
        const int k0_ = ((s) & 15) * KT;                                           \
        _Pragma("unroll")                                                          \
        for (int i = 0; i < 4; ++i)                                                \
            rA[i] = *(const float4*)(As_ + (size_t)(m0 + arow + 32 * i) * DIN +    \
                                     k0_ + akq * 4);                               \
        _Pragma("unroll")                                                          \
        for (int g = 0; g < 4; ++g) {                                              \
            const float* Wg_ = Ws_[g];                                             \
            rB0[g] = *(const float4*)(Wg_ + (size_t)(k0_ + 2 * kp) * DH +          \
                                      n0 + nq * 4);                                \
            rB1[g] = *(const float4*)(Wg_ + (size_t)(k0_ + 2 * kp + 1) * DH +      \
                                      n0 + nq * 4);                                \
        }                                                                          \
    } while (0)

    ISSUE(0);

    for (int s = 0; s < NSTEP; ++s) {
        // ---- packed convert + store staged tile to LDS
        #pragma unroll
        for (int i = 0; i < 4; ++i) {
            h2* dst = (h2*)&lA[arow + 32 * i][akq * 4];
            dst[0] = pkh(rA[i].x, rA[i].y);
            dst[1] = pkh(rA[i].z, rA[i].w);
        }
        #pragma unroll
        for (int g = 0; g < 4; ++g) {
            *(h2*)&lB[g][nq * 4 + 0][2 * kp] = pkh(rB0[g].x, rB1[g].x);
            *(h2*)&lB[g][nq * 4 + 1][2 * kp] = pkh(rB0[g].y, rB1[g].y);
            *(h2*)&lB[g][nq * 4 + 2][2 * kp] = pkh(rB0[g].z, rB1[g].z);
            *(h2*)&lB[g][nq * 4 + 3][2 * kp] = pkh(rB0[g].w, rB1[g].w);
        }
        __syncthreads();

        // ---- prefetch next step's global tiles while WMMAs run
        if (s + 1 < NSTEP) ISSUE(s + 1);

        // ---- A fragment (16x32 f16), reused by all 16 WMMAs
        const int mrow = wave * 16 + nl;
        const v8h alo = *(const v8h*)&lA[mrow][khalf * 16];
        const v8h ahi = *(const v8h*)&lA[mrow][khalf * 16 + 8];
        v16h a;
        #pragma unroll
        for (int e = 0; e < 8; ++e) { a[e] = alo[e]; a[e + 8] = ahi[e]; }

        // ---- software-pipelined B fragments: load t+1 before consuming t
        v16h bcur;
        {
            const v8h blo = *(const v8h*)&lB[0][nl][khalf * 16];
            const v8h bhi = *(const v8h*)&lB[0][nl][khalf * 16 + 8];
            #pragma unroll
            for (int e = 0; e < 8; ++e) { bcur[e] = blo[e]; bcur[e + 8] = bhi[e]; }
        }
        #pragma unroll
        for (int t = 0; t < 16; ++t) {
            v16h bnext;
            if (t < 15) {
                const int g  = (t + 1) >> 2;
                const int nn = ((t + 1) & 3) * 16 + nl;
                const v8h blo = *(const v8h*)&lB[g][nn][khalf * 16];
                const v8h bhi = *(const v8h*)&lB[g][nn][khalf * 16 + 8];
                #pragma unroll
                for (int e = 0; e < 8; ++e) { bnext[e] = blo[e]; bnext[e + 8] = bhi[e]; }
            }
            acc[t] = __builtin_amdgcn_wmma_f32_16x16x32_f16(
                false, a, false, bcur, (short)0, acc[t], false, false);
            if (t < 15) bcur = bnext;
        }
        __syncthreads();
    }
#undef ISSUE

    // ---- fused epilogue: bias + gates + state update, write h_t (f32)
    #pragma unroll
    for (int ns = 0; ns < 4; ++ns) {
        const int ng  = n0 + ns * 16 + nl;
        const float vbf = bf[ng], vbi = bi[ng], vbo = bo[ng], vbc = bc[ng];
        #pragma unroll
        for (int r = 0; r < 8; ++r) {
            const int mg = m0 + wave * 16 + khalf * 8 + r;  // C layout: VGPR r -> row
            const float pf = acc[0 * 4 + ns][r] + vbf;
            const float pi = acc[1 * 4 + ns][r] + vbi;
            const float po = acc[2 * 4 + ns][r] + vbo;
            const float pc = acc[3 * 4 + ns][r] + vbc;
            const float ft = sigmoid_fast(pf);
            const float it = sigmoid_fast(pi);
            const float ot = sigmoid_fast(po);
            const float ch = tanh_fast(pc);
            const float cp = Cprev[(size_t)mg * DH + ng];
            const float ct = ft * cp + it * ch;
            H[(size_t)mg * DH + ng] = ot * tanh_fast(ct);
        }
    }
}

extern "C" void kernel_launch(void* const* d_in, const int* in_sizes, int n_in,
                              void* d_out, int out_size, void* d_ws, size_t ws_size,
                              hipStream_t stream) {
    const float* X  = (const float*)d_in[0];
    const float* Cp = (const float*)d_in[1];
    const float* Wf = (const float*)d_in[2];
    const float* Wi = (const float*)d_in[3];
    const float* Wo = (const float*)d_in[4];
    const float* Wc = (const float*)d_in[5];
    const float* Uf = (const float*)d_in[6];
    const float* Ui = (const float*)d_in[7];
    const float* Uo = (const float*)d_in[8];
    const float* Uc = (const float*)d_in[9];
    const float* bf = (const float*)d_in[10];
    const float* bi = (const float*)d_in[11];
    const float* bo = (const float*)d_in[12];
    const float* bc = (const float*)d_in[13];
    float* H = (float*)d_out;

    dim3 grid(DH / NT, BROWS / MT);   // (8, 128)
    dim3 block(256);
    lstm_fused_wmma<<<grid, block, 0, stream>>>(X, Cp, Wf, Wi, Wo, Wc,
                                                Uf, Ui, Uo, Uc,
                                                bf, bi, bo, bc, H);
}
